// DiffMoeMLP_34617436406188
// MI455X (gfx1250) — matrix operand order repaired and checked by reading, not compiled, not run on hardware
//
#include <hip/hip_runtime.h>

typedef unsigned short u16;
typedef unsigned int   u32;
typedef unsigned long long u64;

#define D_DIM  1024
#define NEXP   8
#define DD_DIM 4096
#define BSZ    8192
#define KCAP   1024   // BS * TRAIN_CAP / N_EXP
#define BM 128
#define BN 128
#define BK 32
#define LP 40         // LDS row pitch in u16 (80 bytes: 16B-aligned, spreads banks)

typedef __attribute__((ext_vector_type(8)))  __bf16 bf16x8;
typedef __attribute__((ext_vector_type(16))) __bf16 bf16x16;
typedef __attribute__((ext_vector_type(8)))  float  f32x8;
typedef __attribute__((ext_vector_type(4)))  unsigned int u32x4;
typedef __attribute__((ext_vector_type(8)))  int          i32x8;
typedef __attribute__((ext_vector_type(4)))  int          i32x4;

__device__ __forceinline__ u16 f2bf(float f) {
  u32 u = __builtin_bit_cast(u32, f);
  u += 0x7fffu + ((u >> 16) & 1u);            // round-to-nearest-even
  return (u16)(u >> 16);
}
__device__ __forceinline__ float bf2f(u16 h) {
  u32 u = ((u32)h) << 16;
  return __builtin_bit_cast(float, u);
}
__device__ __forceinline__ float gelu_tanh(float x) {
  return 0.5f * x * (1.0f + tanhf(0.7978845608028654f * (x + 0.044715f * x * x * x)));
}
__device__ __forceinline__ float wave_sum(float v) {
#pragma unroll
  for (int m = 16; m >= 1; m >>= 1) v += __shfl_xor(v, m, 32);
  return v;
}
__device__ __forceinline__ bf16x16 cat16(bf16x8 lo, bf16x8 hi) {
  return __builtin_shufflevector(lo, hi, 0,1,2,3,4,5,6,7,8,9,10,11,12,13,14,15);
}

// ---------------------------------------------------------------------------
// TDM: async-load a 128x32 bf16 tile (row pitch aPitch elements) from global
// into LDS at ldsOff, inserting 16B padding per 64B row => LP=40 u16 pitch.
// D# per CDNA5 ISA (2-D tensor, groups 2/3 zero).
// Toolchain is the 6-arg builtin: (u32x4, i32x8, i32x4, i32x4, i32x8, i32 cpol)
// ---------------------------------------------------------------------------
__device__ __forceinline__ void tdm_load_tile_bf16(const u16* gsrc, u32 ldsOff, long aPitch)
{
  u64 ga = (u64)(uintptr_t)gsrc;              // byte address, fits 57 bits
  u32x4 g0;
  g0.x = 1u;                                  // count=1, is_restore=0, gather=0
  g0.y = ldsOff;                              // lds_addr (bytes)
  g0.z = (u32)ga;                             // global_addr[31:0]
  g0.w = (u32)((ga >> 32) & 0x01FFFFFFu) | 0x80000000u;  // addr[56:32] | type=2
  i32x8 g1;
  // data_size=1(2B), pad_enable, pad_interval=3(16 dwords), pad_amount=3(4 dwords)
  g1[0] = (1 << 16) | (1 << 20) | (3 << 22) | (3 << 25);
  g1[1] = (int)0xFFFF0000u;                   // tensor_dim0 = 0x7FFFFFFF (low16<<16)
  g1[2] = (int)(0x00007FFFu | 0xFFFF0000u);   // tensor_dim0 hi | tensor_dim1 lo
  g1[3] = (int)(0x00007FFFu | (32u << 16));   // tensor_dim1 hi | tile_dim0=32
  g1[4] = 128;                                // tile_dim1=128, tile_dim2=0
  g1[5] = (int)(u32)aPitch;                   // tensor_dim0_stride (elements)
  g1[6] = 0;                                  // stride hi | tensor_dim1_stride lo
  g1[7] = 0;
  i32x4 z4 = {0, 0, 0, 0};
  i32x8 z8 = {0, 0, 0, 0, 0, 0, 0, 0};
  __builtin_amdgcn_tensor_load_to_lds(g0, g1, z4, z4, z8, 0);
}

// ---------------------------------------------------------------------------
// Shared 128x128x(K) bf16 WMMA GEMM core.  C[m][n] = sum_k A[m][k]*B_T[n][k]
// B is a weight matrix stored (N,K) row-major (i.e. the "W" of y @ W^T).
// A is either f32 (converted on load) or bf16 in workspace (loaded via TDM).
// ---------------------------------------------------------------------------
template<bool AF32>
__device__ __forceinline__ void gemm_core(
    const void* __restrict__ Av, long aOff, long aPitch,
    const float* __restrict__ B, long bOff, int K,
    int M0, int N0, u16* As, u16* Bs, f32x8 (&acc)[2][4])
{
  const int tid  = threadIdx.x;
  const int lane = tid & 31, wv = tid >> 5;
  const int mW = (wv & 3) * 32, nW = (wv >> 2) * 64;
  const int rA  = lane & 15;
  const int kLA = (lane < 16) ? 0 : 8;    // A-frag K base (ISA 16-bit A layout)
  const int kLB = (lane < 16) ? 0 : 8;    // B-frag K base (ISA 16-bit B layout)
  const u32 ldsAs = (u32)(uintptr_t)As;   // generic LDS ptr: low 32 = LDS offset

  for (int k0 = 0; k0 < K; k0 += BK) {
    // ---- stage A tile (128 x 32) into LDS as bf16 ----
    if constexpr (AF32) {
      const float* A = (const float*)Av;
#pragma unroll
      for (int i = 0; i < 4; ++i) {
        int lin = tid * 4 + i;                    // 0..1023 float4 chunks
        int r = lin >> 3, c = (lin & 7) * 4;
        float4 v = *(const float4*)(A + aOff + (long)(M0 + r) * aPitch + k0 + c);
        ushort4 o; o.x = f2bf(v.x); o.y = f2bf(v.y); o.z = f2bf(v.z); o.w = f2bf(v.w);
        *(ushort4*)(As + r * LP + c) = o;
      }
    } else {
      if (wv == 0) {                              // one wave drives the TDM
        const u16* A = (const u16*)Av;
        tdm_load_tile_bf16(A + aOff + (long)M0 * aPitch + k0, ldsAs, aPitch);
      }
    }
    // ---- stage B tile (128 x 32): rows of W are K-contiguous ----
#pragma unroll
    for (int i = 0; i < 4; ++i) {
      int lin = tid * 4 + i;
      int r = lin >> 3, c = (lin & 7) * 4;
      float4 v = *(const float4*)(B + bOff + (long)(N0 + r) * (long)K + k0 + c);
      ushort4 o; o.x = f2bf(v.x); o.y = f2bf(v.y); o.z = f2bf(v.z); o.w = f2bf(v.w);
      *(ushort4*)(Bs + r * LP + c) = o;
    }
    if constexpr (!AF32) {
      if (wv == 0) __builtin_amdgcn_s_wait_tensorcnt(0);
    }
    __syncthreads();

    // ---- build fragments and issue 8 WMMAs ----
    bf16x16 af[2], bfm[4];
#pragma unroll
    for (int im = 0; im < 2; ++im) {
      int row = mW + im * 16 + rA;
      bf16x8 lo = *(const bf16x8*)(As + row * LP + kLA);
      bf16x8 hi = *(const bf16x8*)(As + row * LP + kLA + 16);
      af[im] = cat16(lo, hi);
    }
#pragma unroll
    for (int jn = 0; jn < 4; ++jn) {
      int col = nW + jn * 16 + rA;
      bf16x8 lo = *(const bf16x8*)(Bs + col * LP + kLB);
      bf16x8 hi = *(const bf16x8*)(Bs + col * LP + kLB + 16);
      bfm[jn] = cat16(lo, hi);
    }
#pragma unroll
    for (int im = 0; im < 2; ++im)
#pragma unroll
      for (int jn = 0; jn < 4; ++jn)
        acc[im][jn] = __builtin_amdgcn_wmma_f32_16x16x32_bf16(
            false, af[im], false, bfm[jn], (short)0, acc[im][jn], false, false);
    __syncthreads();
  }
}

// ---------------------------------------------------------------------------
// GEMM + bias + gelu -> bf16 output (cap-predictor fc1 and expert fc1)
// ---------------------------------------------------------------------------
template<bool AF32>
__global__ __launch_bounds__(256) void gemm_bias_gelu_kernel(
    const void* __restrict__ Av, long aZOff, long aPitch,
    const float* __restrict__ B, long bZOff,
    const float* __restrict__ bias, long biasZOff,
    u16* __restrict__ C, long cZOff, int cPitch, int K)
{
  __shared__ __align__(16) u16 As[BM * LP];
  __shared__ __align__(16) u16 Bs[BN * LP];
  const int z  = blockIdx.z;
  const int M0 = blockIdx.y * BM, N0 = blockIdx.x * BN;
  f32x8 acc[2][4] = {};
  gemm_core<AF32>(Av, aZOff * z, aPitch, B, bZOff * z, K, M0, N0, As, Bs, acc);

  const int lane = threadIdx.x & 31, wv = threadIdx.x >> 5;
  const int mW = (wv & 3) * 32, nW = (wv >> 2) * 64;
  const int half = (lane < 16) ? 0 : 8;
#pragma unroll
  for (int im = 0; im < 2; ++im) {
#pragma unroll
    for (int jn = 0; jn < 4; ++jn) {
      int col  = N0 + nW + jn * 16 + (lane & 15);
      float bv = bias[biasZOff * z + col];
      int rowB = M0 + mW + im * 16 + half;
#pragma unroll
      for (int r = 0; r < 8; ++r) {
        float v = gelu_tanh(acc[im][jn][r] + bv);
        C[cZOff * z + (long)(rowB + r) * cPitch + col] = f2bf(v);
      }
    }
  }
}

// ---------------------------------------------------------------------------
// Expert fc2: GEMM + b2 + *kept_w, atomically scattered into out[kept_idx]
// ---------------------------------------------------------------------------
__global__ __launch_bounds__(256) void gemm_fc2_scatter_kernel(
    const u16* __restrict__ H1, const float* __restrict__ fc2s,
    const float* __restrict__ b2s, const float* __restrict__ kw,
    const int* __restrict__ kidx, float* __restrict__ out)
{
  __shared__ __align__(16) u16 As[BM * LP];
  __shared__ __align__(16) u16 Bs[BN * LP];
  const int z  = blockIdx.z;
  const int M0 = blockIdx.y * BM, N0 = blockIdx.x * BN;
  f32x8 acc[2][4] = {};
  gemm_core<false>(H1, (long)KCAP * DD_DIM * z, DD_DIM,
                   fc2s, (long)D_DIM * DD_DIM * z, DD_DIM,
                   M0, N0, As, Bs, acc);

  const int lane = threadIdx.x & 31, wv = threadIdx.x >> 5;
  const int mW = (wv & 3) * 32, nW = (wv >> 2) * 64;
  const int half = (lane < 16) ? 0 : 8;
#pragma unroll
  for (int im = 0; im < 2; ++im) {
#pragma unroll
    for (int jn = 0; jn < 4; ++jn) {
      int col  = N0 + nW + jn * 16 + (lane & 15);
      float bv = b2s[(long)z * D_DIM + col];
      int rowB = M0 + mW + im * 16 + half;
#pragma unroll
      for (int r = 0; r < 8; ++r) {
        int m = rowB + r;                               // expert-local row 0..1023
        float w = kw[m * NEXP + z];
        int tok = kidx[m * NEXP + z];
        atomicAdd(out + (long)tok * D_DIM + col, (acc[im][jn][r] + bv) * w);
      }
    }
  }
}

// ---------------------------------------------------------------------------
// out = x (f32 copy) + zero the loss accumulator
// ---------------------------------------------------------------------------
__global__ void copy_init_kernel(const float4* __restrict__ x, float4* __restrict__ out,
                                 float* __restrict__ lossAcc)
{
  long i = (long)blockIdx.x * blockDim.x + threadIdx.x;
  out[i] = x[i];
  if (i == 0) *lossAcc = 0.0f;
}

// ---------------------------------------------------------------------------
// Gate: scores[m][e] = 0.5*(tanh(x[m] . Wg[e]) + 1)   (1 wave per row)
// ---------------------------------------------------------------------------
__global__ __launch_bounds__(256) void gate_kernel(
    const float* __restrict__ x, const float* __restrict__ Wg, float* __restrict__ scores)
{
  const int lane = threadIdx.x & 31;
  const int m = blockIdx.x * 8 + (threadIdx.x >> 5);
  const float* xr = x + (long)m * D_DIM;
  float acc[NEXP] = {};
  for (int i = lane; i < D_DIM; i += 32) {
    float xv = xr[i];
#pragma unroll
    for (int e = 0; e < NEXP; ++e) acc[e] += xv * Wg[e * D_DIM + i];
  }
#pragma unroll
  for (int e = 0; e < NEXP; ++e) acc[e] = wave_sum(acc[e]);
  if (lane == 0) {
#pragma unroll
    for (int e = 0; e < NEXP; ++e)
      scores[(long)m * NEXP + e] = 0.5f * (tanhf(acc[e]) + 1.0f);
  }
}

// ---------------------------------------------------------------------------
// Per-expert top-k via in-LDS bitonic sort of (score,idx) packed u64 keys.
// Stable-argsort tie order preserved via (N-1-i) in low bits.  Also builds mask.
// ---------------------------------------------------------------------------
__global__ __launch_bounds__(1024) void topk_kernel(
    const float* __restrict__ scores, float* __restrict__ kept_w,
    int* __restrict__ kept_idx, float* __restrict__ mask)
{
  extern __shared__ u64 keys[];              // 8192 * 8B = 64 KB
  const int e = blockIdx.x, tid = threadIdx.x;
  const int N = BSZ;
  for (int i = tid; i < N; i += 1024) {
    mask[(long)i * NEXP + e] = 0.0f;
    u32 u = __builtin_bit_cast(u32, scores[(long)i * NEXP + e]);
    u ^= (u >> 31) ? 0xFFFFFFFFu : 0x80000000u;     // orderable bits
    keys[i] = ((u64)u << 32) | (u32)(N - 1 - i);    // tie-break: smaller idx wins
  }
  __syncthreads();
  for (int k = 2; k <= N; k <<= 1) {
    for (int j = k >> 1; j > 0; j >>= 1) {
      for (int i = tid; i < N; i += 1024) {
        int ij = i ^ j;
        if (ij > i) {
          u64 a = keys[i], b = keys[ij];
          bool desc = ((i & k) == 0);
          bool sw = desc ? (a < b) : (a > b);       // overall descending sort
          if (sw) { keys[i] = b; keys[ij] = a; }
        }
      }
      __syncthreads();
    }
  }
  if (tid < KCAP) {
    u64 kv = keys[tid];
    u32 u = (u32)(kv >> 32);
    u = (u & 0x80000000u) ? (u ^ 0x80000000u) : ~u;
    int idx = N - 1 - (int)(kv & 0xFFFFFFFFu);
    kept_w[tid * NEXP + e]   = __builtin_bit_cast(float, u);
    kept_idx[tid * NEXP + e] = idx;
    mask[(long)idx * NEXP + e] = 1.0f;
  }
}

// ---------------------------------------------------------------------------
// Gather selected rows + LayerNorm -> bf16 (1 wave per output row)
// ---------------------------------------------------------------------------
__global__ __launch_bounds__(256) void ln_gather_kernel(
    const float* __restrict__ x, const int* __restrict__ kidx,
    const float* __restrict__ g, const float* __restrict__ b, u16* __restrict__ yb)
{
  const int lane = threadIdx.x & 31;
  const int j = blockIdx.x * 8 + (threadIdx.x >> 5);
  const int t = kidx[j];
  const float* xr = x + (long)t * D_DIM;
  float s = 0.f, sq = 0.f;
  for (int i = lane; i < D_DIM; i += 32) { float v = xr[i]; s += v; sq += v * v; }
  s = wave_sum(s); sq = wave_sum(sq);
  float mu  = s * (1.0f / D_DIM);
  float var = sq * (1.0f / D_DIM) - mu * mu;
  float inv = rsqrtf(var + 1e-5f);
  u16* yr = yb + (long)j * D_DIM;
  for (int i = lane; i < D_DIM; i += 32)
    yr[i] = f2bf((xr[i] - mu) * inv * g[i] + b[i]);
}

// ---------------------------------------------------------------------------
// logits = h @ cp_w2^T + b2 ; loss += logaddexp(0,z) - z*mask  (1 wave per row)
// ---------------------------------------------------------------------------
__global__ __launch_bounds__(256) void cap_loss_kernel(
    const u16* __restrict__ h, const float* __restrict__ w2,
    const float* __restrict__ b2, const float* __restrict__ mask,
    float* __restrict__ lossAcc)
{
  const int lane = threadIdx.x & 31;
  const int m = blockIdx.x * 8 + (threadIdx.x >> 5);
  const u16* hr = h + (long)m * D_DIM;
  float acc[NEXP] = {};
  for (int i = lane; i < D_DIM; i += 32) {
    float hv = bf2f(hr[i]);
#pragma unroll
    for (int e = 0; e < NEXP; ++e) acc[e] += hv * w2[e * D_DIM + i];
  }
#pragma unroll
  for (int e = 0; e < NEXP; ++e) acc[e] = wave_sum(acc[e]);
  if (lane == 0) {
    float L = 0.f;
#pragma unroll
    for (int e = 0; e < NEXP; ++e) {
      float z = acc[e] + b2[e];
      L += fmaxf(z, 0.f) + log1pf(expf(-fabsf(z))) - z * mask[(long)m * NEXP + e];
    }
    atomicAdd(lossAcc, L);
  }
}

__global__ void finalize_kernel(const float* __restrict__ lossAcc, float* __restrict__ out_loss)
{
  *out_loss = *lossAcc * (1.0f / (float)(BSZ * NEXP));
}

// ---------------------------------------------------------------------------
extern "C" void kernel_launch(void* const* d_in, const int* in_sizes, int n_in,
                              void* d_out, int out_size, void* d_ws, size_t ws_size,
                              hipStream_t stream) {
  (void)in_sizes; (void)n_in; (void)out_size; (void)ws_size;
  const float* x     = (const float*)d_in[0];
  const float* Wg    = (const float*)d_in[1];
  const float* cp_w1 = (const float*)d_in[2];
  const float* cp_b1 = (const float*)d_in[3];
  const float* cp_w2 = (const float*)d_in[4];
  const float* cp_b2 = (const float*)d_in[5];
  const float* ln_g  = (const float*)d_in[6];
  const float* ln_b  = (const float*)d_in[7];
  const float* fc1s  = (const float*)d_in[8];
  const float* b1s   = (const float*)d_in[9];
  const float* fc2s  = (const float*)d_in[10];
  const float* b2s   = (const float*)d_in[11];
  float* out = (float*)d_out;

  // workspace layout (bytes, 16B-aligned slices)
  char* ws = (char*)d_ws;
  float* scores   = (float*)(ws + 0);                    // 8192*8*4   = 262144
  float* kept_w   = (float*)(ws + 262144);               // 1024*8*4   = 32768
  int*   kept_idx = (int*)  (ws + 295040);               // 1024*8*4   = 32768
  float* maskp    = (float*)(ws + 327808);               // 8192*8*4   = 262144
  float* lossAcc  = (float*)(ws + 590080);               // 4 (+pad)
  u16*   hcap     = (u16*)  (ws + 590336);               // 8192*1024*2 = 16 MB
  u16*   yb       = (u16*)  (ws + 590336 + 16777216);    // 8192*1024*2 = 16 MB
  u16*   h1       = (u16*)  (ws + 590336 + 2*16777216);  // 8*1024*4096*2 = 64 MB

  // 1) out = x ; loss = 0
  copy_init_kernel<<<(BSZ * D_DIM / 4) / 256, 256, 0, stream>>>(
      (const float4*)x, (float4*)out, lossAcc);

  // 2) gate scores
  gate_kernel<<<BSZ / 8, 256, 0, stream>>>(x, Wg, scores);

  // 3) per-expert top-k (bitonic, 64KB dynamic LDS per block)
  topk_kernel<<<NEXP, 1024, BSZ * sizeof(u64), stream>>>(scores, kept_w, kept_idx, maskp);

  // 4) gather + layernorm -> bf16
  ln_gather_kernel<<<BSZ / 8, 256, 0, stream>>>(x, kept_idx, ln_g, ln_b, yb);

  // 5) capacity predictor hidden: h = gelu(x @ cp_w1^T + cp_b1)   [8192 x 1024]
  gemm_bias_gelu_kernel<true><<<dim3(D_DIM / BN, BSZ / BM, 1), 256, 0, stream>>>(
      x, 0L, (long)D_DIM, cp_w1, 0L, cp_b1, 0L, hcap, 0L, D_DIM, D_DIM);

  // 6) cap logits + logaddexp loss
  cap_loss_kernel<<<BSZ / 8, 256, 0, stream>>>(hcap, cp_w2, cp_b2, maskp, lossAcc);

  // 7) expert fc1: h1[n] = gelu(Y_n @ fc1s[n]^T + b1s[n])   [8 x 1024 x 4096]
  //    A tiles via TDM (bf16 in ws), B tiles converted f32->bf16 on the fly.
  gemm_bias_gelu_kernel<false><<<dim3(DD_DIM / BN, KCAP / BM, NEXP), 256, 0, stream>>>(
      yb, (long)D_DIM, (long)(NEXP * D_DIM),
      fc1s, (long)DD_DIM * D_DIM,
      b1s, (long)DD_DIM,
      h1, (long)KCAP * DD_DIM, DD_DIM, D_DIM);

  // 8) expert fc2 + scale + scatter-add into out (A tiles via TDM)
  gemm_fc2_scatter_kernel<<<dim3(D_DIM / BN, KCAP / BM, NEXP), 256, 0, stream>>>(
      h1, fc2s, b2s, kept_w, kept_idx, out);

  // 9) normalize loss into d_out tail
  finalize_kernel<<<1, 1, 0, stream>>>(lossAcc, out + (long)BSZ * D_DIM);
}
